// MoEBasicBlock_15058155340547
// MI455X (gfx1250) — matrix operand order repaired and checked by reading, not compiled, not run on hardware
//
#include <hip/hip_runtime.h>

// ---------------------------------------------------------------------------
// Types / helpers
// ---------------------------------------------------------------------------
typedef __attribute__((ext_vector_type(16))) __bf16 v16bf;
typedef __attribute__((ext_vector_type(8)))  float  v8f;

union FragBF16 { unsigned int u[8]; v16bf v; };

static __device__ __forceinline__ __bf16 bits_to_bf16(unsigned short s) {
    union { unsigned short s; __bf16 h; } u; u.s = s; return u.h;
}
static __device__ __forceinline__ __bf16 f32_to_bf16(float f) {
    union { float f; unsigned int u; } x; x.f = f;
    unsigned int r = x.u + 0x7FFFu + ((x.u >> 16) & 1u);   // round-to-nearest-even
    return bits_to_bf16((unsigned short)(r >> 16));
}
static __device__ __forceinline__ __bf16 to_bf16(float v)  { return f32_to_bf16(v); }
static __device__ __forceinline__ __bf16 to_bf16(__bf16 v) { return v; }

#define NB   16
#define NC   256
#define NH   64
#define NW   64
#define NE   512
#define HW   4096                    // NH*NW
#define NIMG 16777216                // NB*NC*NH*NW

// ---------------------------------------------------------------------------
// Gate kernel: gate = softmax(emb @ W + b) over channels. One block per batch.
// ---------------------------------------------------------------------------
__global__ void gates_kernel(const float* __restrict__ emb,
                             const float* __restrict__ w,
                             const float* __restrict__ bias,
                             float* __restrict__ gout) {
    __shared__ float red[NC];
    const int b = blockIdx.x, c = threadIdx.x;
    float acc = bias[c];
    const float* e = emb + b * NE;
    for (int k = 0; k < NE; ++k) acc += e[k] * w[k * NC + c];
    red[c] = acc; __syncthreads();
    for (int s = 128; s > 0; s >>= 1) {
        if (c < s) red[c] = fmaxf(red[c], red[c + s]);
        __syncthreads();
    }
    const float mx = red[0]; __syncthreads();
    const float ex = __expf(acc - mx);
    red[c] = ex; __syncthreads();
    for (int s = 128; s > 0; s >>= 1) {
        if (c < s) red[c] += red[c + s];
        __syncthreads();
    }
    gout[b * NC + c] = ex / red[0];
}

// ---------------------------------------------------------------------------
// Weight repack: f32 [OC,IC,3,3] -> bf16 [tap][OC][IC]
// ---------------------------------------------------------------------------
__global__ void repack_w(const float* __restrict__ w1, const float* __restrict__ w2,
                         __bf16* __restrict__ o1, __bf16* __restrict__ o2) {
    const int n = 9 * NC * NC;
    int i = blockIdx.x * blockDim.x + threadIdx.x;
    if (i >= 2 * n) return;
    const float* w = (i < n) ? w1 : w2;
    __bf16*      o = (i < n) ? o1 : o2;
    const int j  = (i < n) ? i : i - n;
    const int t  = j / (NC * NC);
    const int r  = j % (NC * NC);
    const int oc = r / NC, ic = r % NC;
    o[j] = f32_to_bf16(w[(oc * NC + ic) * 9 + t]);
}

// ---------------------------------------------------------------------------
// Implicit-GEMM 3x3 conv with bf16 WMMA, f32 accum, gate fused in epilogue.
// Grid: x = B*H (one (b,h) output row of 64 px), y = 4 (OC tiles of 64).
// Block: 256 threads = 8 waves; wave (wi,wj) owns a 16-OC x 32-px strip
// (two 16x16 WMMA tiles, 2 WMMAs per staged K-chunk of 32).
// ---------------------------------------------------------------------------
template <typename Tin>
__global__ void __launch_bounds__(256)
conv3x3_wmma(const Tin* __restrict__ in, const __bf16* __restrict__ wbf,
             const float* __restrict__ gate, float* __restrict__ y) {
    __shared__ __bf16 As[64 * 32];   // [oc_local][ic]   (A operand tile)
    __shared__ __bf16 Bs[64 * 32];   // [px_local][ic]   (B operand tile, px-major)

    const int b      = blockIdx.x >> 6;
    const int h      = blockIdx.x & 63;
    const int ocbase = blockIdx.y << 6;
    const int tid    = threadIdx.x;
    const int wv = tid >> 5, lane = tid & 31;
    const int wi = wv >> 1, wj = wv & 1;
    const int half = lane >> 4, mn = lane & 15;

    // per-thread staging coordinates (shared by A and B fills)
    const int frow = tid >> 2;              // 0..63 : oc_local for A / out w for B
    const int fseg = (tid & 3) << 3;        // 0,8,16,24 : ic sub-chunk

    v8f acc0 = {}; v8f acc1 = {};

    for (int t = 0; t < 9; ++t) {
        const int dh = t / 3 - 1, dw = t % 3 - 1;
        const int hh = h + dh;
        const bool rowOK = ((unsigned)hh < (unsigned)NH);
        const int  hc    = rowOK ? hh : 0;          // clamped (block-uniform)
        const int  wsrc  = frow + dw;
        const int  wc    = min(max(wsrc, 0), NW - 1);
        const bool ok    = rowOK && (wsrc == wc);   // in-bounds source pixel
        // prefetch next tap's weight tile (global_prefetch_b8)
        if (t < 8 && tid == 0)
            __builtin_prefetch(wbf + (((t + 1) * NC + ocbase) << 8), 0, 1);

        for (int kc = 0; kc < 8; ++kc) {
            const int icb = kc << 5;
            // ---- stage A: 64 oc x 32 ic, one 16B copy per thread ----
            *(uint4*)&As[(frow << 5) + fseg] =
                *(const uint4*)(wbf + ((t * NC + ocbase + frow) << 8) + icb + fseg);
            // ---- stage B: 64 px x 32 ic (px-major), branch-free ----
            {
                const Tin* src = in + ((((b << 8) + icb + fseg) << 12) + (hc << 6) + wc);
                __bf16 tmp[8];
                #pragma unroll
                for (int j = 0; j < 8; ++j) {
                    const __bf16 v = to_bf16(src[j << 12]);      // stride = H*W
                    tmp[j] = ok ? v : bits_to_bf16(0);
                }
                *(uint4*)&Bs[(frow << 5) + fseg] = *(const uint4*)tmp;
            }
            __syncthreads();

            // ---- fragments per documented CDNA5 16-bit layouts ----
            FragBF16 a, b0, b1;
            #pragma unroll
            for (int v = 0; v < 8; ++v) {
                // A 16x32: lane half selects K+8 / K+24 groups; K pairs contiguous
                const int ka = (v < 4) ? (2 * v + 8 * half) : (16 + 2 * (v - 4) + 8 * half);
                a.u[v] = *(const unsigned int*)&As[((wi * 16 + mn) << 5) + ka];
                // B 32x16: same K order as A (lane half selects K+8 / K+24 groups); K pairs contiguous
                const int kb = (v < 4) ? (2 * v + 8 * half) : (16 + 2 * (v - 4) + 8 * half);
                const int c0 = (wj << 5) + mn;
                b0.u[v] = *(const unsigned int*)&Bs[(c0 << 5) + kb];
                b1.u[v] = *(const unsigned int*)&Bs[((c0 + 16) << 5) + kb];
            }
            acc0 = __builtin_amdgcn_wmma_f32_16x16x32_bf16(
                       false, a.v, false, b0.v, (short)0, acc0, false, false);
            acc1 = __builtin_amdgcn_wmma_f32_16x16x32_bf16(
                       false, a.v, false, b1.v, (short)0, acc1, false, false);
            __syncthreads();
        }
    }

    // ---- epilogue: D layout VGPR r -> M = r + 8*half, N = lane&15 ----
    #pragma unroll
    for (int r = 0; r < 8; ++r) {
        const int ocl = wi * 16 + r + 8 * half;
        const int oc  = ocbase + ocl;
        const float gd = gate[(b << 8) + oc];
        const int c0 = (wj << 5) + mn;
        float* dst = y + (((b << 8) + oc) << 12) + (h << 6);
        dst[c0]      = acc0[r] * gd;
        dst[c0 + 16] = acc1[r] * gd;
    }
}

// ---------------------------------------------------------------------------
// Per-channel BN statistics: mean + rsqrt(var + eps). One block per channel.
// float4-vectorized reads (channel plane is 4096 floats = 1024 float4).
// ---------------------------------------------------------------------------
__global__ void bnstats(const float* __restrict__ y,
                        float* __restrict__ mean, float* __restrict__ inv) {
    __shared__ float rs[256], rq[256];
    const int c = blockIdx.x, tid = threadIdx.x;
    const float4* y4 = (const float4*)y;
    float s = 0.f, q = 0.f;
    for (int i = tid; i < (NB * HW) / 4; i += 256) {
        const int b = i >> 10, p = i & 1023;
        const float4 v = y4[(((b << 8) + c) << 10) + p];
        s += v.x + v.y + v.z + v.w;
        q += v.x * v.x + v.y * v.y + v.z * v.z + v.w * v.w;
    }
    rs[tid] = s; rq[tid] = q; __syncthreads();
    for (int st = 128; st > 0; st >>= 1) {
        if (tid < st) { rs[tid] += rs[tid + st]; rq[tid] += rq[tid + st]; }
        __syncthreads();
    }
    if (tid == 0) {
        const float m   = rs[0] * (1.f / (NB * HW));
        const float var = rq[0] * (1.f / (NB * HW)) - m * m;
        mean[c] = m;
        inv[c]  = rsqrtf(var + 1e-5f);
    }
}

// ---------------------------------------------------------------------------
// BN + ReLU + convert to bf16 (feeds conv2's B operand). float4 in, 4x bf16 out.
// ---------------------------------------------------------------------------
__global__ void bn_relu_bf16(const float* __restrict__ y,
                             const float* __restrict__ mean, const float* __restrict__ inv,
                             const float* __restrict__ g, const float* __restrict__ be,
                             __bf16* __restrict__ a) {
    const int i = blockIdx.x * blockDim.x + threadIdx.x;   // float4 index
    if (i >= NIMG / 4) return;
    const int c = (i >> 10) & 255;
    const float sc = inv[c] * g[c];
    const float sh = be[c] - mean[c] * sc;
    const float4 v = ((const float4*)y)[i];
    __bf16 o[4];
    o[0] = f32_to_bf16(fmaxf(v.x * sc + sh, 0.f));
    o[1] = f32_to_bf16(fmaxf(v.y * sc + sh, 0.f));
    o[2] = f32_to_bf16(fmaxf(v.z * sc + sh, 0.f));
    o[3] = f32_to_bf16(fmaxf(v.w * sc + sh, 0.f));
    ((uint2*)a)[i] = *(const uint2*)o;
}

// ---------------------------------------------------------------------------
// Final: BN + residual + ReLU -> f32 output. float4 vectorized.
// ---------------------------------------------------------------------------
__global__ void bn_add_relu(const float* __restrict__ y,
                            const float* __restrict__ mean, const float* __restrict__ inv,
                            const float* __restrict__ g, const float* __restrict__ be,
                            const float* __restrict__ x, float* __restrict__ out) {
    const int i = blockIdx.x * blockDim.x + threadIdx.x;   // float4 index
    if (i >= NIMG / 4) return;
    const int c = (i >> 10) & 255;
    const float sc = inv[c] * g[c];
    const float sh = be[c] - mean[c] * sc;
    const float4 v = ((const float4*)y)[i];
    const float4 r = ((const float4*)x)[i];
    float4 o;
    o.x = fmaxf(v.x * sc + sh + r.x, 0.f);
    o.y = fmaxf(v.y * sc + sh + r.y, 0.f);
    o.z = fmaxf(v.z * sc + sh + r.z, 0.f);
    o.w = fmaxf(v.w * sc + sh + r.w, 0.f);
    ((float4*)out)[i] = o;
}

// ---------------------------------------------------------------------------
// Launcher
// ---------------------------------------------------------------------------
extern "C" void kernel_launch(void* const* d_in, const int* in_sizes, int n_in,
                              void* d_out, int out_size, void* d_ws, size_t ws_size,
                              hipStream_t stream) {
    const float* x    = (const float*)d_in[0];
    const float* emb  = (const float*)d_in[1];
    const float* w1   = (const float*)d_in[2];
    const float* w2   = (const float*)d_in[3];
    const float* g1w  = (const float*)d_in[4];
    const float* g1b  = (const float*)d_in[5];
    const float* g2w  = (const float*)d_in[6];
    const float* g2b  = (const float*)d_in[7];
    const float* bn1g = (const float*)d_in[8];
    const float* bn1b = (const float*)d_in[9];
    const float* bn2g = (const float*)d_in[10];
    const float* bn2b = (const float*)d_in[11];

    float* out   = (float*)d_out;         // [16,256,64,64]
    float* gate1 = out + NIMG;            // [16,256]
    float* gate2 = gate1 + NB * NC;       // [16,256]

    // workspace layout (256B-aligned offsets), ~103 MB total
    char* ws = (char*)d_ws;
    __bf16* w1bf = (__bf16*)(ws);                                   // 1,179,648 B
    __bf16* w2bf = (__bf16*)(ws + 1179648);                         // 1,179,648 B
    __bf16* a1   = (__bf16*)(ws + 2359296);                         // 33,554,432 B
    float*  ybuf = (float*)(ws + 2359296 + 33554432);               // 67,108,864 B
    float*  st   = (float*)(ws + 2359296 + 33554432 + 67108864);    // 4 KB
    float *mean1 = st, *inv1 = st + 256, *mean2 = st + 512, *inv2 = st + 768;

    // 1) gates
    gates_kernel<<<NB, NC, 0, stream>>>(emb, g1w, g1b, gate1);
    gates_kernel<<<NB, NC, 0, stream>>>(emb, g2w, g2b, gate2);
    // 2) weight repack to bf16 [tap][oc][ic]
    repack_w<<<(2 * 9 * NC * NC + 255) / 256, 256, 0, stream>>>(w1, w2, w1bf, w2bf);
    // 3) conv1 (f32 input) * gate1 -> ybuf
    conv3x3_wmma<float><<<dim3(NB * NH, 4), 256, 0, stream>>>(x, w1bf, gate1, ybuf);
    // 4) BN1 stats, BN1+ReLU -> a1 (bf16)
    bnstats<<<NC, 256, 0, stream>>>(ybuf, mean1, inv1);
    bn_relu_bf16<<<NIMG / 4 / 256, 256, 0, stream>>>(ybuf, mean1, inv1, bn1g, bn1b, a1);
    // 5) conv2 (bf16 input) * gate2 -> ybuf (reused)
    conv3x3_wmma<__bf16><<<dim3(NB * NH, 4), 256, 0, stream>>>(a1, w2bf, gate2, ybuf);
    // 6) BN2 stats, BN2 + residual + ReLU -> out
    bnstats<<<NC, 256, 0, stream>>>(ybuf, mean2, inv2);
    bn_add_relu<<<NIMG / 4 / 256, 256, 0, stream>>>(ybuf, mean2, inv2, bn2g, bn2b, x, out);
}
